// GCU_Pyrimad_17824114278456
// MI455X (gfx1250) — matrix operand order repaired and checked by reading, not compiled, not run on hardware
//
#include <hip/hip_runtime.h>
#include <cstdint>

// ---------------------------------------------------------------------------
// MI455X (gfx1250) implementation.
// Compute-bound (~175 GFLOP vs ~5us HBM floor) => route the two big GEMM
// families (3x3 convs via implicit GEMM, fused reproj+ext 1x1) through
// v_wmma_f32_16x16x32_bf16 (f32 accumulate). Everything else is tiny VALU.
// Conv kernel: double-buffered LDS software pipeline, division-free K-chunk
// bookkeeping (incremental ky/kx/ic0), vectorized NCHW epilogue.
// Workspace usage ~75 MB (assumed ws_size is at least that).
// ---------------------------------------------------------------------------

typedef __attribute__((ext_vector_type(16))) __bf16 v16bf;
typedef __attribute__((ext_vector_type(8)))  float  v8f;

__device__ __forceinline__ unsigned short f2bf(float x) {
  union { float f; uint32_t u; } v; v.f = x;
  uint32_t r = v.u + 0x7FFFu + ((v.u >> 16) & 1u);   // round-to-nearest-even
  return (unsigned short)(r >> 16);
}

// ---------------- layout converters -----------------------------------------

// NCHW f32 [2][C][4096] -> NHWC bf16 [2][4096][C]
__global__ __launch_bounds__(256) void k_cvt_nhwc(const float* __restrict__ in,
                                                  unsigned short* __restrict__ out,
                                                  int C) {
  size_t idx = (size_t)blockIdx.x * 256 + threadIdx.x;
  size_t total = (size_t)2 * C * 4096;
  if (idx >= total) return;
  int pix = (int)(idx & 4095);
  size_t r = idx >> 12;
  int c = (int)(r % C);
  int b = (int)(r / C);
  out[((size_t)b * 4096 + pix) * C + c] = f2bf(in[idx]);
}

// OIHW f32 [OC][IC][9] -> [OC][9][IC] bf16 (K-chunk = 32 contiguous ic)
__global__ __launch_bounds__(256) void k_cvt_w(const float* __restrict__ w,
                                               unsigned short* __restrict__ out,
                                               int OC, int IC) {
  size_t idx = (size_t)blockIdx.x * 256 + threadIdx.x;
  size_t total = (size_t)OC * 9 * IC;
  if (idx >= total) return;
  int ic = (int)(idx % IC);
  int kp = (int)((idx / IC) % 9);
  int oc = (int)(idx / ((size_t)IC * 9));
  out[idx] = f2bf(w[((size_t)oc * IC + ic) * 9 + kp]);
}

// ---------------- implicit-GEMM 3x3 conv (WMMA bf16) -------------------------
// Block: 256 thr = 8 waves. Tile: 128 pixels x 64 oc, K-chunks of 32 (ic).
// Double-buffered LDS: A (im2col) [2][128][40], B (weights) [2][64][40];
// padded stride 40 halfwords (80B: 16B-aligned, conflict-free 20-dword rows).
// Software pipeline: prefetch chunk c+1 from global into registers while the
// WMMAs consume chunk c from LDS; one workgroup barrier per K-chunk.

#define LDSTR 40
#define ABUF  (128 * LDSTR)
#define BBUF  (64 * LDSTR)

__global__ __launch_bounds__(256) void conv3x3_igemm(
    const unsigned short* __restrict__ in_nhwc,  // [2][4096][IC] bf16
    const unsigned short* __restrict__ w_r,      // [OC][9][IC]  bf16
    float* __restrict__ out,                     // nhwc [2][4096][OC] or nchw [2][OC][4096]
    int IC, int OC, int out_nchw) {
  const int b    = blockIdx.z;
  const int mt   = blockIdx.x;          // 32 pixel tiles of 128
  const int nt   = blockIdx.y;          // oc tiles of 64
  const int t    = threadIdx.x;
  const int wave = t >> 5, lane = t & 31;

  __shared__ unsigned short As[2 * ABUF];
  __shared__ unsigned short Bs[2 * BBUF];

  v8f acc[4];
#pragma unroll
  for (int i = 0; i < 4; ++i)
#pragma unroll
    for (int j = 0; j < 8; ++j) acc[i][j] = 0.f;

  const unsigned short* inb = in_nhwc + (size_t)b * 4096 * IC;
  const int chunks = (IC >> 5) * 9;

  // A loader coords: 2 threads/pixel, 32B each
  const int pa = t >> 1;
  const int ha = (t & 1) * 16;
  const int py = (mt * 128 + pa) >> 6;
  const int px = (mt * 128 + pa) & 63;
  // B loader coords: 4 threads/oc row, 16B each
  const int ocl = t >> 2;
  const int qb  = (t & 3) * 8;
  const unsigned short* wrow = w_r + (size_t)(nt * 64 + ocl) * 9 * IC + qb;

  const int adw  = (lane >> 4) << 2;    // A frag dword offset: +0 / +4 (K +0/+8)
  const int dbB  = (lane >> 4) << 3;    // B frag dword base:   +0 / +8 (K +0/+16)
  const int mrow = (wave << 4) + (lane & 15);

  // incremental K-chunk coordinates (no divisions in the hot loop)
  int ky = 0, kx = 0, ic0 = 0, kpIC = 0;   // kpIC = (ky*3+kx)*IC
  uint4 ra0, ra1, rb;

  auto prefetch = [&]() {
    int iy = py + ky - 1, ix = px + kx - 1;
    ra0.x = ra0.y = ra0.z = ra0.w = 0u;
    ra1 = ra0;
    if ((unsigned)iy < 64u && (unsigned)ix < 64u) {
      const uint4* src = (const uint4*)(inb + (size_t)(iy * 64 + ix) * IC + ic0 + ha);
      ra0 = src[0]; ra1 = src[1];
    }
    rb = *(const uint4*)(wrow + kpIC + ic0);
  };
  auto advance = [&]() {
    ic0 += 32;
    if (ic0 == IC) { ic0 = 0; kpIC += IC; ++kx; if (kx == 3) { kx = 0; ++ky; } }
  };
  auto stage = [&](int buf) {
    uint4* da = (uint4*)(As + buf * ABUF + pa * LDSTR + ha);
    da[0] = ra0; da[1] = ra1;
    *(uint4*)(Bs + buf * BBUF + ocl * LDSTR + qb) = rb;
  };

  prefetch();
  stage(0);
  __syncthreads();

  for (int c = 0; c < chunks; ++c) {
    const int p = c & 1;
    if (c + 1 < chunks) { advance(); prefetch(); }
    // ---- fragments: A (16x32: M = lane&15; K = 2v+(v>=4?8:0)+8*(lane>>4)),
    //                 B (32x16: N = lane&15; K = 2v+(v>=4?8:0)+8*(lane>>4))
    union { v16bf v; uint32_t u[8]; } fa, fb[4];
    {
      const uint32_t* a32 = (const uint32_t*)(As + p * ABUF + mrow * LDSTR);
#pragma unroll
      for (int v = 0; v < 8; ++v) fa.u[v] = a32[v + ((v >= 4) ? 4 : 0) + adw];
    }
#pragma unroll
    for (int tn = 0; tn < 4; ++tn) {
      const uint32_t* b32 =
          (const uint32_t*)(Bs + p * BBUF + ((tn << 4) + (lane & 15)) * LDSTR);
#pragma unroll
      for (int v = 0; v < 8; ++v) fb[tn].u[v] = b32[v + ((v >= 4) ? 4 : 0) + adw];
    }
#pragma unroll
    for (int tn = 0; tn < 4; ++tn)
      acc[tn] = __builtin_amdgcn_wmma_f32_16x16x32_bf16(
          false, fa.v, false, fb[tn].v, (short)0, acc[tn], false, false);
    if (c + 1 < chunks) {
      stage(1 - p);          // other buffer: last read in the previous iteration, before its barrier
      __syncthreads();
    }
  }

  // ---- epilogue: D layout: N = lane&15; M = v + 8*(lane>>4)
  const int ms = mt * 128 + (wave << 4) + ((lane >> 4) << 3);
  const int nc = lane & 15;
#pragma unroll
  for (int tn = 0; tn < 4; ++tn) {
    int oc = nt * 64 + (tn << 4) + nc;
    if (out_nchw) {
      float* dst = out + ((size_t)b * OC + oc) * 4096 + ms;  // 8 consecutive pixels
      float4 s0 = { acc[tn][0], acc[tn][1], acc[tn][2], acc[tn][3] };
      float4 s1 = { acc[tn][4], acc[tn][5], acc[tn][6], acc[tn][7] };
      *(float4*)(dst)     = s0;
      *(float4*)(dst + 4) = s1;
    } else {
#pragma unroll
      for (int v = 0; v < 8; ++v)
        out[((size_t)b * 4096 + ms + v) * OC + oc] = acc[tn][v];
    }
  }
}

// ---------------- code projection: Q = softmax_k(-0.5*sum_d r^2) -------------
// one wave per pixel; lane = codeword k. Q padded to stride 32 with zeros.
__global__ __launch_bounds__(256) void k_qsoftmax(
    const float* __restrict__ X,     // [2*4096][256]
    const float* __restrict__ cw,    // [K][256]
    const float* __restrict__ sc,    // [256][K]
    float* __restrict__ Qf,          // [2*4096][32]
    unsigned short* __restrict__ Qb, // [2*4096][32] bf16
    int K) {
  int wave = threadIdx.x >> 5, lane = threadIdx.x & 31;
  size_t gp = (size_t)blockIdx.x * 8 + wave;   // 0..8191
  const float* xrow = X + gp * 256;
  float acc = 0.f;
  if (lane < K) {
    for (int d = 0; d < 256; ++d) {
      float r = (xrow[d] - cw[lane * 256 + d]) * sc[d * K + lane];
      acc += r * r;
    }
  }
  float s = -0.5f * acc;
  float m = (lane < K) ? s : -3.4e38f;
  for (int off = 16; off; off >>= 1) m = fmaxf(m, __shfl_xor(m, off));
  float e = (lane < K) ? __expf(s - m) : 0.f;
  float sum = e;
  for (int off = 16; off; off >>= 1) sum += __shfl_xor(sum, off);
  float q = e / sum;
  Qf[gp * 32 + lane] = q;
  Qb[gp * 32 + lane] = f2bf(q);
}

// S1[b][k] = sum_n Q
__global__ __launch_bounds__(256) void k_s1(const float* __restrict__ Qf,
                                            float* __restrict__ S1) {
  int b = blockIdx.x >> 5, k = blockIdx.x & 31, tid = threadIdx.x;
  float a = 0.f;
  for (int n = tid; n < 4096; n += 256) a += Qf[((size_t)b * 4096 + n) * 32 + k];
  __shared__ float red[256];
  red[tid] = a; __syncthreads();
  for (int s = 128; s; s >>= 1) { if (tid < s) red[tid] += red[tid + s]; __syncthreads(); }
  if (!tid) S1[b * 32 + k] = red[0];
}

// SX[b][d][k] = sum_n X[b,n,d] * Q[b,n,k]
__global__ __launch_bounds__(256) void k_sx(const float* __restrict__ X,
                                            const float* __restrict__ Qf,
                                            float* __restrict__ SX) {
  int b = blockIdx.x >> 8, d = blockIdx.x & 255, tid = threadIdx.x;
  float acc[32];
#pragma unroll
  for (int k = 0; k < 32; ++k) acc[k] = 0.f;
  for (int n = tid; n < 4096; n += 256) {
    float xv = X[((size_t)b * 4096 + n) * 256 + d];
    const float* qr = Qf + ((size_t)b * 4096 + n) * 32;
#pragma unroll
    for (int k = 0; k < 32; ++k) acc[k] += xv * qr[k];
  }
  __shared__ float red[256];
#pragma unroll
  for (int k = 0; k < 32; ++k) {
    red[tid] = acc[k]; __syncthreads();
    for (int s = 128; s; s >>= 1) { if (tid < s) red[tid] += red[tid + s]; __syncthreads(); }
    if (!tid) SX[((size_t)b * 256 + d) * 32 + k] = red[0];
    __syncthreads();
  }
}

// Z_[d,k] = sc*(SX - cw*S1)/S1, then L2-normalize over d.  One block per (b,k).
__global__ __launch_bounds__(256) void k_zmake(const float* __restrict__ SX,
                                               const float* __restrict__ S1,
                                               const float* __restrict__ cw,
                                               const float* __restrict__ sc,
                                               float* __restrict__ Z, int K) {
  int b = blockIdx.x / K, k = blockIdx.x % K, d = threadIdx.x;
  float s1 = S1[b * 32 + k];
  float z = sc[d * K + k] * (SX[((size_t)b * 256 + d) * 32 + k] - cw[k * 256 + d] * s1) / s1;
  __shared__ float red[256];
  red[d] = z * z; __syncthreads();
  for (int s = 128; s; s >>= 1) { if (d < s) red[d] += red[d + s]; __syncthreads(); }
  float inv = rsqrtf(red[0]);
  Z[((size_t)b * 256 + d) * 32 + k] = z * inv;
}

// ---------------- graph conv (tiny) -----------------------------------------
__global__ __launch_bounds__(256) void k_adj(const float* __restrict__ Z,
                                             float* __restrict__ adj, int K) {
  int b = blockIdx.x;
  for (int idx = threadIdx.x; idx < K * K; idx += 256) {
    int k = idx / K, l = idx % K;
    float a = 0.f;
    for (int d = 0; d < 256; ++d)
      a += Z[((size_t)b * 256 + d) * 32 + k] * Z[((size_t)b * 256 + d) * 32 + l];
    adj[b * 1024 + k * 32 + l] = a;
  }
}

__global__ __launch_bounds__(256) void k_support(const float* __restrict__ gcnw,
                                                 const float* __restrict__ Z,
                                                 float* __restrict__ sup, int K) {
  int idx = blockIdx.x * 256 + threadIdx.x;
  if (idx >= 2 * 256 * K) return;
  int b = idx / (256 * K), d = (idx / K) % 256, k = idx % K;
  float a = 0.f;
  for (int j = 0; j < 256; ++j)
    a += gcnw[d * 256 + j] * Z[((size_t)b * 256 + j) * 32 + k];
  sup[((size_t)b * 256 + d) * 32 + k] = a;
}

__global__ __launch_bounds__(256) void k_gout(const float* __restrict__ sup,
                                              const float* __restrict__ adj,
                                              float* __restrict__ gout, int K) {
  int idx = blockIdx.x * 256 + threadIdx.x;
  if (idx >= 2 * 256 * K) return;
  int b = idx / (256 * K), d = (idx / K) % 256, l = idx % K;
  float a = 0.f;
  for (int k = 0; k < K; ++k)
    a += sup[((size_t)b * 256 + d) * 32 + k] * adj[b * 1024 + k * 32 + l];
  gout[((size_t)b * 256 + d) * 32 + l] = fmaxf(a, 0.f);
}

// BN over (b,l) per channel d; fold gamma/beta into scale/shift
__global__ __launch_bounds__(256) void k_bnstat(const float* __restrict__ gout,
                                                const float* __restrict__ g,
                                                const float* __restrict__ bt,
                                                float* __restrict__ bnsc,
                                                float* __restrict__ bnsh, int K) {
  int d = threadIdx.x;
  float s = 0.f, s2 = 0.f;
  for (int b = 0; b < 2; ++b)
    for (int l = 0; l < K; ++l) {
      float v = gout[((size_t)b * 256 + d) * 32 + l];
      s += v; s2 += v * v;
    }
  float cnt = (float)(2 * K);
  float mean = s / cnt, var = s2 / cnt - mean * mean;
  float sc = g[d] * rsqrtf(var + 1e-5f);
  bnsc[d] = sc;
  bnsh[d] = bt[d] - mean * sc;
}

// EWG[b][oc][32] = ext_w @ BN(gout), bf16, zero-padded past K
__global__ __launch_bounds__(256) void k_ewg(const float* __restrict__ ext_w,
                                             const float* __restrict__ gout,
                                             const float* __restrict__ bnsc,
                                             const float* __restrict__ bnsh,
                                             unsigned short* __restrict__ EWG, int K) {
  int idx = blockIdx.x * 256 + threadIdx.x;
  if (idx >= 2 * 512 * 32) return;
  int b = idx / (512 * 32), oc = (idx / 32) % 512, k = idx % 32;
  float a = 0.f;
  if (k < K) {
    for (int d = 0; d < 256; ++d)
      a += ext_w[oc * 256 + d] *
           (bnsc[d] * gout[((size_t)b * 256 + d) * 32 + k] + bnsh[d]);
  }
  EWG[idx] = f2bf(a);
}

// ---------------- fused reproj + ext 1x1 + residual (WMMA, K-dim = 32) -------
// out[pix][oc] = Q[pix,:] . EWG[oc,:] + c[b,oc,pix];  bf16 into concat buffer.
__global__ __launch_bounds__(256) void reproj_ext_wmma(
    const unsigned short* __restrict__ Qb,   // [2][4096][32] bf16
    const unsigned short* __restrict__ EWG,  // [2][512][32]  bf16
    const float* __restrict__ cres,          // [2][512][4096] f32 NCHW
    unsigned short* __restrict__ cc,         // [2][4096][1536] bf16
    int choff) {
  int b = blockIdx.y;
  int t = threadIdx.x, wave = t >> 5, lane = t & 31;
  int strip = blockIdx.x * 128 + wave * 16;
  // A frag from Q rows (M = lane&15): two 16B segments at K0, K0+16
  union { v16bf v; uint4 q[2]; } fa;
  {
    const unsigned short* arow = Qb + ((size_t)b * 4096 + strip + (lane & 15)) * 32;
    int k0 = (lane >> 4) << 3;
    fa.q[0] = *(const uint4*)(arow + k0);
    fa.q[1] = *(const uint4*)(arow + k0 + 16);
  }
  const int nc = lane & 15;
  const int ms = strip + ((lane >> 4) << 3);
  const int kb = (lane >> 4) << 3;
  for (int ntile = 0; ntile < 32; ++ntile) {
    union { v16bf v; uint4 q[2]; } fb;
    int oc = ntile * 16 + nc;
    const unsigned short* brow = EWG + ((size_t)b * 512 + oc) * 32;
    fb.q[0] = *(const uint4*)(brow + kb);
    fb.q[1] = *(const uint4*)(brow + kb + 16);
    v8f acc;
#pragma unroll
    for (int j = 0; j < 8; ++j) acc[j] = 0.f;
    acc = __builtin_amdgcn_wmma_f32_16x16x32_bf16(
        false, fa.v, false, fb.v, (short)0, acc, false, false);
#pragma unroll
    for (int v = 0; v < 8; ++v) {
      int pix = ms + v;
      float val = acc[v] + cres[((size_t)b * 512 + oc) * 4096 + pix];
      cc[((size_t)b * 4096 + pix) * 1536 + choff + oc] = f2bf(val);
    }
  }
}

// ---------------- final BN2 + ReLU -------------------------------------------
__global__ __launch_bounds__(256) void k_bn2stat(const float* __restrict__ sm,
                                                 const float* __restrict__ g,
                                                 const float* __restrict__ bt,
                                                 float* __restrict__ sc,
                                                 float* __restrict__ sh) {
  int oc = blockIdx.x, tid = threadIdx.x;
  float s = 0.f, s2 = 0.f;
  for (int b = 0; b < 2; ++b)
    for (int n = tid; n < 4096; n += 256) {
      float v = sm[((size_t)b * 512 + oc) * 4096 + n];
      s += v; s2 += v * v;
    }
  __shared__ float r1[256], r2[256];
  r1[tid] = s; r2[tid] = s2; __syncthreads();
  for (int st = 128; st; st >>= 1) {
    if (tid < st) { r1[tid] += r1[tid + st]; r2[tid] += r2[tid + st]; }
    __syncthreads();
  }
  if (!tid) {
    float mean = r1[0] / 8192.f, var = r2[0] / 8192.f - mean * mean;
    float k = g[oc] * rsqrtf(var + 1e-5f);
    sc[oc] = k;
    sh[oc] = bt[oc] - mean * k;
  }
}

__global__ __launch_bounds__(256) void k_bn2apply(const float* __restrict__ sm,
                                                  const float* __restrict__ sc,
                                                  const float* __restrict__ sh,
                                                  float* __restrict__ out) {
  size_t idx = (size_t)blockIdx.x * 256 + threadIdx.x;
  if (idx >= (size_t)2 * 512 * 4096) return;
  int oc = (int)((idx >> 12) & 511);
  out[idx] = fmaxf(sc[oc] * sm[idx] + sh[oc], 0.f);
}

// ---------------------------------------------------------------------------
extern "C" void kernel_launch(void* const* d_in, const int* in_sizes, int n_in,
                              void* d_out, int out_size, void* d_ws, size_t ws_size,
                              hipStream_t stream) {
  (void)in_sizes; (void)n_in; (void)out_size; (void)ws_size;
  const float* c3 = (const float*)d_in[0];
  const float* c4 = (const float*)d_in[1];
  const float* c5 = (const float*)d_in[2];
  const float* smooth_w = (const float*)d_in[24];
  const float* smg = (const float*)d_in[25];
  const float* smb = (const float*)d_in[26];

  char* wsb = (char*)d_ws;
  size_t off = 0;
  auto alloc = [&](size_t bytes) {
    void* p = wsb + off;
    off += (bytes + 255) & ~(size_t)255;
    return p;
  };
  unsigned short* chw  = (unsigned short*)alloc((size_t)2 * 4096 * 512 * 2);   // NHWC bf16 input
  unsigned short* wr   = (unsigned short*)alloc((size_t)512 * 9 * 1536 * 2);   // reordered weights
  float*          X    = (float*)alloc((size_t)2 * 4096 * 256 * 4);            // conv out NHWC
  float*          Qf   = (float*)alloc((size_t)2 * 4096 * 32 * 4);
  unsigned short* Qb   = (unsigned short*)alloc((size_t)2 * 4096 * 32 * 2);
  float*          S1   = (float*)alloc(2 * 32 * 4);
  float*          SX   = (float*)alloc((size_t)2 * 256 * 32 * 4);
  float*          Z    = (float*)alloc((size_t)2 * 256 * 32 * 4);
  float*          adj  = (float*)alloc((size_t)2 * 32 * 32 * 4);
  float*          sup  = (float*)alloc((size_t)2 * 256 * 32 * 4);
  float*          gout = (float*)alloc((size_t)2 * 256 * 32 * 4);
  float*          bnsc = (float*)alloc(256 * 4);
  float*          bnsh = (float*)alloc(256 * 4);
  unsigned short* EWG  = (unsigned short*)alloc((size_t)2 * 512 * 32 * 2);
  unsigned short* cc   = (unsigned short*)alloc((size_t)2 * 4096 * 1536 * 2);  // concat NHWC bf16
  float*          sm   = (float*)alloc((size_t)2 * 512 * 4096 * 4);            // smooth out NCHW
  float*          b2sc = (float*)alloc(512 * 4);
  float*          b2sh = (float*)alloc(512 * 4);

  struct Br { const float* cin; int K; int base; int choff; };
  // concat order is [x_8, x_16, x_32] along channels
  Br brs[3] = { { c3, 32, 3, 1024 }, { c4, 16, 10, 512 }, { c5, 8, 17, 0 } };

  for (int i = 0; i < 3; ++i) {
    const Br& br = brs[i];
    const float* conv_w = (const float*)d_in[br.base + 0];
    const float* cw     = (const float*)d_in[br.base + 1];
    const float* sc     = (const float*)d_in[br.base + 2];
    const float* gcnw   = (const float*)d_in[br.base + 3];
    const float* bng    = (const float*)d_in[br.base + 4];
    const float* bnb    = (const float*)d_in[br.base + 5];
    const float* ext_w  = (const float*)d_in[br.base + 6];

    { size_t tot = (size_t)2 * 512 * 4096;
      k_cvt_nhwc<<<dim3((unsigned)((tot + 255) / 256)), 256, 0, stream>>>(br.cin, chw, 512); }
    { size_t tot = (size_t)256 * 9 * 512;
      k_cvt_w<<<dim3((unsigned)((tot + 255) / 256)), 256, 0, stream>>>(conv_w, wr, 256, 512); }
    conv3x3_igemm<<<dim3(32, 4, 2), 256, 0, stream>>>(chw, wr, X, 512, 256, 0);
    k_qsoftmax<<<dim3(1024), 256, 0, stream>>>(X, cw, sc, Qf, Qb, br.K);
    k_s1<<<dim3(64), 256, 0, stream>>>(Qf, S1);
    k_sx<<<dim3(512), 256, 0, stream>>>(X, Qf, SX);
    k_zmake<<<dim3(2 * br.K), 256, 0, stream>>>(SX, S1, cw, sc, Z, br.K);
    k_adj<<<dim3(2), 256, 0, stream>>>(Z, adj, br.K);
    { int tot = 2 * 256 * br.K;
      k_support<<<dim3((tot + 255) / 256), 256, 0, stream>>>(gcnw, Z, sup, br.K);
      k_gout<<<dim3((tot + 255) / 256), 256, 0, stream>>>(sup, adj, gout, br.K); }
    k_bnstat<<<dim3(1), 256, 0, stream>>>(gout, bng, bnb, bnsc, bnsh, br.K);
    k_ewg<<<dim3(128), 256, 0, stream>>>(ext_w, gout, bnsc, bnsh, EWG, br.K);
    reproj_ext_wmma<<<dim3(32, 2), 256, 0, stream>>>(Qb, EWG, br.cin, cc, br.choff);
  }

  // smooth conv 1536->512 on the concat, then BN2 + ReLU
  { size_t tot = (size_t)512 * 9 * 1536;
    k_cvt_w<<<dim3((unsigned)((tot + 255) / 256)), 256, 0, stream>>>(smooth_w, wr, 512, 1536); }
  conv3x3_igemm<<<dim3(32, 8, 2), 256, 0, stream>>>(cc, wr, sm, 1536, 512, 1);
  k_bn2stat<<<dim3(512), 256, 0, stream>>>(sm, smg, smb, b2sc, b2sh);
  { size_t tot = (size_t)2 * 512 * 4096;
    k_bn2apply<<<dim3((unsigned)((tot + 255) / 256)), 256, 0, stream>>>(sm, b2sc, b2sh, (float*)d_out); }
}
